// BPTT_MultiTask_35631048688158
// MI455X (gfx1250) — compile-verified, matrix operand order checked by reading, not run on hardware
//
#include <hip/hip_runtime.h>

// CDNA5 / gfx1250 persistent-RNN kernel, round 3.
// One workgroup: 512 threads = 16 wave32s on one WGP, runs all T steps.
// J = G + M (bf16, padded 500->512 = 512KB) split across the WGP:
//   - wave w owns rows [32w, 32w+32) as two 16-row blocks
//   - k-chunks 0..7  : register-resident WMMA B-tiles (16 x v16bf / wave)
//   - k-chunks 8..15 : LDS-resident B-tiles in operand layout (256 KB)
// Per step: r_t (bf16, LDS, double-buffered) -> A operand (identical rows),
// 32 v_wmma_f32_16x16x32_bf16 per wave on two alternating accumulators,
// SOFTWARE-PIPELINED one k-chunk ahead so ds_load latency overlaps WMMA
// (partial s_wait_dscnt instead of full drains), one s_barrier per step.

typedef __attribute__((ext_vector_type(16))) __bf16        v16bf;
typedef __attribute__((ext_vector_type(8)))  float         v8f;
typedef __attribute__((ext_vector_type(4)))  unsigned int  v4u;

#define NPAD   512
#define NKC    16      // k-chunks of 32
#define NKREG  8       // k-chunks 0..7 held in registers
#define ALPHA  0.1f    // DT/TAU

// bf16 pair (J[row][col], J[row][col+1]) packed into one dword, zero padded.
__device__ __forceinline__ unsigned int pack_pair(const float* __restrict__ G,
                                                  const float* __restrict__ M,
                                                  int N, int row, int col) {
    float a0 = 0.f, a1 = 0.f;
    if (row < N) {
        long b = (long)row * N;
        if (col     < N) a0 = G[b + col]     + M[b + col];
        if (col + 1 < N) a1 = G[b + col + 1] + M[b + col + 1];
    }
    union { __bf16 h[2]; unsigned int u; } p;
    p.h[0] = (__bf16)a0;
    p.h[1] = (__bf16)a1;
    return p.u;
}

__global__ __launch_bounds__(512)
void rnn_persistent_wmma(const float* __restrict__ x0,
                         const float* __restrict__ G,
                         const float* __restrict__ M,
                         const float* __restrict__ Wro,
                         const int*   __restrict__ Tptr,
                         float*       __restrict__ out,
                         int N)
{
    __shared__ float        rf[2][NPAD];           // r_t in f32 (readout)
    __shared__ unsigned int rbuf[2][NPAD / 2];     // r_t in packed bf16 (A)
    __shared__ float        wro[3 * NPAD];         // zero-padded W_ro
    // LDS B-tiles: [wave][rb][kc-8][lane][8 dwords] = 16*2*8*32*8 dwords
    __shared__ unsigned int ldsB[16 * 2 * 8 * 32 * 8];   // 256 KB

    const int tid = threadIdx.x;
    const int w   = tid >> 5;          // wave 0..15 -> rows [32w, 32w+32)
    const int L   = tid & 31;
    const int n   = L & 15;
    const int hi  = L >> 4;
    const int T   = *Tptr;

    // ---- stage W_ro (zero padded) ----
    for (int idx = tid; idx < 3 * NPAD; idx += 512) {
        int c = idx >> 9, j = idx & (NPAD - 1);
        wro[idx] = (j < N) ? Wro[c * N + j] : 0.f;
    }

    // 16-bit operand layout (ISA 7.12.2): dword vv of a tile holds K-pair
    //   k0 = (vv<4 ? 2vv : 8+2vv) + hi*8
    const int row0 = w * 32 + n;          // row-block 0 row for this lane
    const int row1 = row0 + 16;           // row-block 1 row

    // ---- register-resident B-tiles: k-chunks 0..7 ----
    v16bf jt0[NKREG], jt1[NKREG];
    #pragma unroll
    for (int kc = 0; kc < NKREG; ++kc) {
        union { unsigned int u[8]; v16bf v; } t0, t1;
        #pragma unroll
        for (int vv = 0; vv < 8; ++vv) {
            int k0  = ((vv < 4) ? 2 * vv : 8 + 2 * vv) + hi * 8;
            int col = kc * 32 + k0;
            t0.u[vv] = pack_pair(G, M, N, row0, col);
            t1.u[vv] = pack_pair(G, M, N, row1, col);
        }
        jt0[kc] = t0.v;
        jt1[kc] = t1.v;
    }

    // ---- LDS-resident B-tiles: k-chunks 8..15 ----
    // dword index: (w*2+i)*2048 + kc8*256 + L*8 + vv
    {
        unsigned int* pBw = &ldsB[w * 4096 + L * 8];
        #pragma unroll 1
        for (int i = 0; i < 2; ++i) {
            int row = i ? row1 : row0;
            #pragma unroll 1
            for (int kc8 = 0; kc8 < 8; ++kc8) {
                unsigned int* dst = pBw + i * 2048 + kc8 * 256;
                #pragma unroll
                for (int vv = 0; vv < 8; ++vv) {
                    int k0  = ((vv < 4) ? 2 * vv : 8 + 2 * vv) + hi * 8;
                    int col = (8 + kc8) * 32 + k0;
                    dst[vv] = pack_pair(G, M, N, row, col);
                }
            }
        }
    }

    // ---- initial state ----
    float xa = 0.f, xb = 0.f;
    if (L < 16) {
        xa = (row0 < N) ? x0[row0] : 0.f;
        xb = (row1 < N) ? x0[row1] : 0.f;
        float ra = (row0 < N) ? tanhf(xa) : 0.f;
        float rb = (row1 < N) ? tanhf(xb) : 0.f;
        rf[0][row0] = ra;  ((__bf16*)rbuf[0])[row0] = (__bf16)ra;
        rf[0][row1] = rb;  ((__bf16*)rbuf[0])[row1] = (__bf16)rb;
    }
    __syncthreads();

    const unsigned int* pB = &ldsB[w * 4096 + L * 8];

    // ---- time loop: one barrier per step, double-buffered r ----
    for (int t = 0; t < T; ++t) {
        const int buf = t & 1, nb = buf ^ 1;
        const unsigned int* rbb = rbuf[buf];

        auto loadA = [&](int kc) -> v16bf {
            union { v4u q[2]; v16bf v; } a;
            a.q[0] = *(const v4u*)&rbb[kc * 16 +     hi * 4];
            a.q[1] = *(const v4u*)&rbb[kc * 16 + 8 + hi * 4];
            return a.v;
        };
        auto loadB = [&](int i, int kc8) -> v16bf {
            union { v4u q[2]; v16bf v; } b;
            const unsigned int* p = pB + i * 2048 + kc8 * 256;
            b.q[0] = *(const v4u*)(p);
            b.q[1] = *(const v4u*)(p + 4);
            return b.v;
        };

        // Software-pipelined matvec: prefetch chunk kc+1 while WMMA-ing kc,
        // so the backend can emit partial s_wait_dscnt instead of 0-drains.
        v8f acc0 = {}, acc1 = {};
        v16bf aCur  = loadA(0);
        v16bf b0Cur = jt0[0];
        v16bf b1Cur = jt1[0];
        #pragma unroll
        for (int kc = 0; kc < NKC; ++kc) {
            v16bf aNxt = aCur, b0Nxt = b0Cur, b1Nxt = b1Cur;
            const int kn = kc + 1;
            if (kn < NKC) {                       // compile-time (unrolled)
                aNxt  = loadA(kn);
                b0Nxt = (kn < NKREG) ? jt0[kn] : loadB(0, kn - NKREG);
                b1Nxt = (kn < NKREG) ? jt1[kn] : loadB(1, kn - NKREG);
            }
            // alternate accumulators -> WMMA RAW hazard distance covered
            acc0 = __builtin_amdgcn_wmma_f32_16x16x32_bf16(
                       false, aCur, false, b0Cur, (short)0, acc0, false, false);
            acc1 = __builtin_amdgcn_wmma_f32_16x16x32_bf16(
                       false, aCur, false, b1Cur, (short)0, acc1, false, false);
            aCur = aNxt;  b0Cur = b0Nxt;  b1Cur = b1Nxt;   // SSA renames
        }
        // Every D VGPR lane = y[rowblock_base + (lane&15)]
        float y0 = acc0[0];
        float y1 = acc1[0];

        // Readout z[t] = W_ro * r_t : waves 0..2, one component each.
        if (w < 3) {
            float p = 0.f;
            #pragma unroll
            for (int ii = 0; ii < 16; ++ii) {
                int j = L + 32 * ii;
                p += wro[w * NPAD + j] * rf[buf][j];
            }
            #pragma unroll
            for (int off = 16; off > 0; off >>= 1)
                p += __shfl_xor(p, off, 32);
            if (L == 0) out[t * 3 + w] = p;
        }

        // Euler update + write r_{t+1} into the other buffer.
        if (L < 16) {
            xa = xa + ALPHA * (y0 - xa);
            xb = xb + ALPHA * (y1 - xb);
            float ra = (row0 < N) ? tanhf(xa) : 0.f;
            float rb = (row1 < N) ? tanhf(xb) : 0.f;
            rf[nb][row0] = ra;  ((__bf16*)rbuf[nb])[row0] = (__bf16)ra;
            rf[nb][row1] = rb;  ((__bf16*)rbuf[nb])[row1] = (__bf16)rb;
        }
        __syncthreads();
    }
}

extern "C" void kernel_launch(void* const* d_in, const int* in_sizes, int n_in,
                              void* d_out, int out_size, void* d_ws, size_t ws_size,
                              hipStream_t stream) {
    const float* x0 = (const float*)d_in[0];
    const float* G  = (const float*)d_in[1];
    const float* M  = (const float*)d_in[2];
    const float* W  = (const float*)d_in[3];
    const int*   T  = (const int*)d_in[4];
    const int N = in_sizes[0];   // 500
    hipLaunchKernelGGL(rnn_persistent_wmma, dim3(1), dim3(512), 0, stream,
                       x0, G, M, W, T, (float*)d_out, N);
}
